// LSTM_22960895165168
// MI455X (gfx1250) — compile-verified, matrix operand order checked by reading, not run on hardware
//
#include <hip/hip_runtime.h>
#include <hip/hip_bf16.h>

// LSTM fused kernel for gfx1250 (MI455X), wave32 + WMMA bf16.
// Grid: 16 blocks (4 batch-groups x 4 N-slice WGs), 512 threads (16 waves).
// Wave w: gate = w>>2, ntile = w&3. Weights register-resident as B-fragments.
// Per step: h A-fragments preloaded up front (L2 latency hidden behind the
// LDS-fed x WMMA chain); x chain software-pipelined; bias folded into the
// first WMMA's C operand. c/h update spread over all 16 waves.

typedef __attribute__((ext_vector_type(16))) __bf16 v16bf;
typedef __attribute__((ext_vector_type(8)))  float  v8f;

#define SEQ    2048
#define BATCH  64
#define EMBED  256
#define HIDDEN 256

// ws layout: [0,256)   : 4 group barrier counters (u32) + pad
//            [256,...) : h double buffer, 4 groups x 2 x 16 x 256 bf16 (64 KB)
#define WS_CNT_OFF  0
#define WS_H_OFF    256
#define WS_U32_N    ((256 + 4 * 2 * 16 * 256 * 2) / 4)

__global__ void lstm_init_ws(uint32_t* __restrict__ ws, int n) {
    int i = blockIdx.x * blockDim.x + threadIdx.x;
    if (i < n) ws[i] = 0u;
}

__global__ __launch_bounds__(512, 1) void lstm_fused(
    const float* __restrict__ embeds,
    const float* __restrict__ Wix, const float* __restrict__ Wih, const float* __restrict__ bi,
    const float* __restrict__ Wfx, const float* __restrict__ Wfh, const float* __restrict__ bf,
    const float* __restrict__ Wgx, const float* __restrict__ Wgh, const float* __restrict__ bg,
    const float* __restrict__ Wox, const float* __restrict__ Woh, const float* __restrict__ bo,
    float* __restrict__ out,
    uint32_t* __restrict__ cnt,          // [4] group barrier counters
    __bf16* __restrict__ hbuf)           // [4][2][16][256]
{
    const int tid   = threadIdx.x;
    const int wave  = tid >> 5;
    const int lane  = tid & 31;
    const int gate  = wave >> 2;          // 0=i 1=f 2=g 3=o
    const int nt    = wave & 3;           // 16-col tile within this WG's 64 cols
    const int nwg   = blockIdx.x & 3;     // N-slice workgroup within group
    const int group = blockIdx.x >> 2;    // batch group (16 rows)
    const int rbase = group * 16;
    const int colbase = nwg * 64 + nt * 16;
    const int lhalf = lane >> 4;          // 0: lanes 0-15, 1: lanes 16-31
    const int lmod  = lane & 15;

    const float* Wx; const float* Wh; const float* bptr;
    if      (gate == 0) { Wx = Wix; Wh = Wih; bptr = bi; }
    else if (gate == 1) { Wx = Wfx; Wh = Wfh; bptr = bf; }
    else if (gate == 2) { Wx = Wgx; Wh = Wgh; bptr = bg; }
    else                { Wx = Wox; Wh = Woh; bptr = bo; }

    // ---- Register-resident B fragments (bf16), 8 K-tiles per matrix ----
    // B tile 32x16: lanes 0-15 -> N=lmod, K = kt*32 + 0..15;
    //               lanes 16-31 -> N=lmod, K = kt*32 + 16..31.
    v16bf WxF[8], WhF[8];
    {
        const int n = colbase + lmod;
        #pragma unroll
        for (int kt = 0; kt < 8; ++kt) {
            const int kb = kt * 32 + lhalf * 16;
            v16bf fx, fh;
            #pragma unroll
            for (int j = 0; j < 16; ++j) {
                fx[j] = (__bf16)Wx[(size_t)(kb + j) * HIDDEN + n];
                fh[j] = (__bf16)Wh[(size_t)(kb + j) * HIDDEN + n];
            }
            WxF[kt] = fx; WhF[kt] = fh;
        }
    }
    // Bias vector, kept resident; used as C operand of the first WMMA.
    v8f bvec;
    {
        const float b = bptr[colbase + lmod];
        #pragma unroll
        for (int r = 0; r < 8; ++r) bvec[r] = b;
    }

    // LDS: gate exchange + double-buffered x in A-fragment order.
    // xbf[buf][kt*32 + lane][16] : lane stride 32B -> conflict-free b128 reads.
    __shared__ float  shg[4][4][16][17];          // 17408 B
    __shared__ __bf16 xbf[2][8 * 32 * 16];        // 16384 B

    // x staging role for this thread: row m, 8 consecutive K values.
    const int sm    = tid >> 5;                   // 0..15  (row)
    const int skblk = (tid & 31) * 8;             // 0..248 (K start, step 8)
    const int skt   = skblk >> 5;
    const int sidx  = ((skt * 32) + ((skblk >> 4) & 1) * 16 + sm) * 16 + (skblk & 15);
    const float* sxbase = embeds + ((size_t)rbase + sm) * EMBED + skblk;

    // Update role: this wave owns rows um0, um0+1 of tile `nt`, col n = lmod.
    const int um0 = gate * 4 + lhalf * 2;
    float c0 = 0.0f, c1 = 0.0f;

    __bf16* hb = hbuf + (size_t)group * (2 * 16 * HIDDEN);

    // ---- Prologue: stage x_0 into xbf[0] ----
    {
        const float* p = sxbase;                  // t = 0
        float f[8];
        #pragma unroll
        for (int j = 0; j < 8; ++j) f[j] = p[j];
        __bf16 b8[8];
        #pragma unroll
        for (int j = 0; j < 8; ++j) b8[j] = (__bf16)f[j];
        *(uint4*)&xbf[0][sidx] = *(uint4*)b8;
    }
    __syncthreads();

    for (int t = 0; t < SEQ; ++t) {
        const int cur = t & 1, nxt = cur ^ 1;
        const __bf16* hcur = hb + (size_t)cur * (16 * HIDDEN);
        __bf16*       hnxt = hb + (size_t)nxt * (16 * HIDDEN);

        // 1) Issue ALL h A-fragment loads up front (L2 latency hidden behind
        //    the x WMMA chain below).
        v16bf ha[8];
        {
            const __bf16* hp = hcur + lmod * HIDDEN + lhalf * 16;
            #pragma unroll
            for (int kt = 0; kt < 8; ++kt)
                ha[kt] = *(const v16bf*)(hp + kt * 32);
        }

        // 2) Kick off global loads of x_{t+1} (overlaps with WMMAs below).
        float fstage[8];
        const bool do_stage = (t + 1 < SEQ);
        if (do_stage) {
            const float* p = sxbase + (size_t)(t + 1) * (BATCH * EMBED);
            #pragma unroll
            for (int j = 0; j < 8; ++j) fstage[j] = p[j];
        }

        // 3) x WMMA chain, software-pipelined LDS fragment loads.
        v8f acc = bvec;
        {
            v16bf acurf = *(const v16bf*)&xbf[cur][(0 * 32 + lane) * 16];
            #pragma unroll
            for (int kt = 0; kt < 8; ++kt) {
                v16bf anext = acurf;
                if (kt < 7)
                    anext = *(const v16bf*)&xbf[cur][((kt + 1) * 32 + lane) * 16];
                acc = __builtin_amdgcn_wmma_f32_16x16x32_bf16(
                    false, acurf, false, WxF[kt], (short)0, acc, false, false);
                acurf = anext;
            }
        }

        // 4) h WMMA chain (fragments already in flight / resident).
        #pragma unroll
        for (int kt = 0; kt < 8; ++kt) {
            acc = __builtin_amdgcn_wmma_f32_16x16x32_bf16(
                false, ha[kt], false, WhF[kt], (short)0, acc, false, false);
        }

        // 5) Convert + store staged x_{t+1} into the other LDS buffer.
        if (do_stage) {
            __bf16 b8[8];
            #pragma unroll
            for (int j = 0; j < 8; ++j) b8[j] = (__bf16)fstage[j];
            *(uint4*)&xbf[nxt][sidx] = *(uint4*)b8;
        }

        // 6) Publish gate pre-activations: C layout -> shg[gate][nt][m][n]
        #pragma unroll
        for (int r = 0; r < 8; ++r)
            shg[gate][nt][r + 8 * lhalf][lmod] = acc[r];
        __syncthreads();

        // 7) All 16 waves: combine gates, update c/h for 2 elements per lane.
        {
            const int n    = lmod;
            const int gcol = nwg * 64 + nt * 16 + n;
            #pragma unroll
            for (int e = 0; e < 2; ++e) {
                const int m = um0 + e;
                const float xi = shg[0][nt][m][n];
                const float xf = shg[1][nt][m][n];
                const float xg = shg[2][nt][m][n];
                const float xo = shg[3][nt][m][n];
                const float i_ = 1.0f / (1.0f + __expf(-xi));
                const float f_ = 1.0f / (1.0f + __expf(-xf));
                const float g_ = tanhf(xg);
                const float o_ = 1.0f / (1.0f + __expf(-xo));
                float& cc = e ? c1 : c0;
                cc = f_ * cc + i_ * g_;
                const float h_ = o_ * tanhf(cc);
                __builtin_nontemporal_store(
                    h_, &out[((size_t)t * BATCH + rbase + m) * HIDDEN + gcol]);
                hnxt[m * HIDDEN + gcol] = (__bf16)h_;
            }
        }
        __threadfence();   // make h visible device-wide before arriving
        __syncthreads();

        // 8) Group barrier across the 4 N-slice WGs (monotonic counter, L2).
        if (tid == 0) {
            __hip_atomic_fetch_add(&cnt[group], 1u,
                                   __ATOMIC_RELEASE, __HIP_MEMORY_SCOPE_AGENT);
            const uint32_t target = 4u * (uint32_t)(t + 1);
            while (__hip_atomic_load(&cnt[group],
                                     __ATOMIC_ACQUIRE, __HIP_MEMORY_SCOPE_AGENT) < target)
                __builtin_amdgcn_s_sleep(1);
        }
        __syncthreads();
    }
}

extern "C" void kernel_launch(void* const* d_in, const int* in_sizes, int n_in,
                              void* d_out, int out_size, void* d_ws, size_t ws_size,
                              hipStream_t stream) {
    (void)in_sizes; (void)n_in; (void)out_size; (void)ws_size;

    const float* embeds = (const float*)d_in[0];
    const float* Wix = (const float*)d_in[1];
    const float* Wih = (const float*)d_in[2];
    const float* bi  = (const float*)d_in[3];
    const float* Wfx = (const float*)d_in[4];
    const float* Wfh = (const float*)d_in[5];
    const float* bf  = (const float*)d_in[6];
    const float* Wgx = (const float*)d_in[7];
    const float* Wgh = (const float*)d_in[8];
    const float* bg  = (const float*)d_in[9];
    const float* Wox = (const float*)d_in[10];
    const float* Woh = (const float*)d_in[11];
    const float* bo  = (const float*)d_in[12];

    uint32_t* ws_u32 = (uint32_t*)d_ws;
    uint32_t* cnt    = (uint32_t*)((char*)d_ws + WS_CNT_OFF);
    __bf16*   hbuf   = (__bf16*)((char*)d_ws + WS_H_OFF);

    // Zero counters + h0 every call (deterministic; ws is poisoned).
    lstm_init_ws<<<(WS_U32_N + 255) / 256, 256, 0, stream>>>(ws_u32, WS_U32_N);

    lstm_fused<<<16, 512, 0, stream>>>(embeds,
                                       Wix, Wih, bi,
                                       Wfx, Wfh, bf,
                                       Wgx, Wgh, bg,
                                       Wox, Woh, bo,
                                       (float*)d_out, cnt, hbuf);
}
